// MultiHeadAttention_71528385347884
// MI455X (gfx1250) — matrix operand order repaired and checked by reading, not compiled, not run on hardware
//
#include <hip/hip_runtime.h>
#include <cstddef>

// ---------- types ----------
typedef __bf16 v16bf  __attribute__((ext_vector_type(16)));
typedef __bf16 bf16x8 __attribute__((ext_vector_type(8)));
typedef float  v8f    __attribute__((ext_vector_type(8)));
typedef int    v4i    __attribute__((ext_vector_type(4)));

#if defined(__AMDGCN__) && __has_builtin(__builtin_amdgcn_global_load_async_to_lds_b128)
#define USE_ASYNC_LDS 1
#else
#define USE_ASYNC_LDS 0
#endif

#if defined(__AMDGCN__)
typedef __attribute__((address_space(1))) v4i gv4i;  // global 16B vector
typedef __attribute__((address_space(3))) v4i lv4i;  // LDS 16B vector
#endif

// 16B global -> LDS copy: async DMA on CDNA5 toolchains, sync fallback otherwise.
__device__ __forceinline__ void cp16_g2l(const __bf16* g, __bf16* l) {
#if USE_ASYNC_LDS
  __builtin_amdgcn_global_load_async_to_lds_b128((gv4i*)g, (lv4i*)l, 0, 0);
#else
  *(bf16x8*)l = *(const bf16x8*)g;
#endif
}

__device__ __forceinline__ void wait_async() {
#if USE_ASYNC_LDS
#if __has_builtin(__builtin_amdgcn_s_wait_asynccnt)
  __builtin_amdgcn_s_wait_asynccnt(0);
#else
  asm volatile("s_wait_asynccnt 0x0" ::: "memory");
#endif
#endif
}

__device__ __forceinline__ void lds_fence() {
  asm volatile("s_wait_dscnt 0x0" ::: "memory");
}

__device__ __forceinline__ v8f wmma_bf16(v16bf a, v16bf b, v8f c) {
  // v_wmma_f32_16x16x32_bf16  (neg_a, A, neg_b, B, c_mod, C, reuse_a, reuse_b)
  return __builtin_amdgcn_wmma_f32_16x16x32_bf16(false, a, false, b, (short)0, c, false, false);
}

// A fragment: 16x32 bf16, row-major source, row stride `stride`.
// lane<16 : row=lane,    K = {0..15};  lane>=16: row=lane-16, K = {16..31}
__device__ __forceinline__ v16bf frag_a(const __bf16* p0, int stride, int lane) {
  const __bf16* p = p0 + (size_t)(lane & 15) * stride + ((lane & 16) ? 16 : 0);
  v16bf f;
#pragma unroll
  for (int e = 0; e < 16; ++e) f[e] = p[e];
  return f;
}

// B fragment: 32x16 bf16 (k x n), k-major source with row stride `stride`.
__device__ __forceinline__ v16bf frag_b(const __bf16* p0, int stride, int lane) {
  const __bf16* p = p0 + (lane & 15) + ((lane & 16) ? (size_t)16 * stride : 0);
  v16bf f;
#pragma unroll
  for (int e = 0; e < 16; ++e) f[e] = p[(size_t)e * stride];
  return f;
}

// B fragment for S = Q*K^T: B[k=dh][n=key] = K[key][dh] -> one 32B contiguous read/lane.
__device__ __forceinline__ v16bf frag_bT(const __bf16* krow0, int stride, int lane) {
  const __bf16* p = krow0 + (size_t)(lane & 15) * stride + ((lane & 16) ? 16 : 0);
  v16bf f;
#pragma unroll
  for (int e = 0; e < 16; ++e) f[e] = p[e];
  return f;
}

// ---------- kernel 1: fp32 -> bf16 convert ----------
__global__ __launch_bounds__(256) void cvt_f32_bf16(const float* __restrict__ s,
                                                    __bf16* __restrict__ d, int n) {
  int i = blockIdx.x * blockDim.x + threadIdx.x;
  int stride = gridDim.x * blockDim.x;
  for (; i < n; i += stride) d[i] = (__bf16)s[i];
}

// ---------- kernels 2/4: 128x128-tile bf16 WMMA GEMM, async double-buffered ----------
struct StageBufs { __bf16 a[2][128][40]; __bf16 b[2][32][136]; };
union SmemU { StageBufs s; float c[128][128]; };

template <int EPI>  // 0: QKV + fused RoPE scatter, 1: plain fp32 store
__global__ __launch_bounds__(256) void gemm_bf16(
    const __bf16* __restrict__ A, const __bf16* __restrict__ Bm,
    int Kdim, int Ncols,
    __bf16* __restrict__ Qh, __bf16* __restrict__ Kh, __bf16* __restrict__ Vh,
    float* __restrict__ Out) {
  __shared__ SmemU sm;
  const int tid = threadIdx.x;
  const int lane = tid & 31;
  const int wid = tid >> 5;
  const int wm = wid >> 1, wn = wid & 1;
  const int hi8 = (lane & 16) ? 8 : 0;
  const int nlo = lane & 15;
  const int m0 = blockIdx.y * 128;
  const int n0 = blockIdx.x * 128;
  const int ar = tid >> 1, ak = (tid & 1) << 4;  // A: 128 rows x 32 k
  const int bk = tid >> 3, bn = (tid & 7) << 4;  // B: 32 k x 128 n

  auto stage = [&](int buf, int kk) {
    const __bf16* ga = A + (size_t)(m0 + ar) * Kdim + kk + ak;
    cp16_g2l(ga, &sm.s.a[buf][ar][ak]);
    cp16_g2l(ga + 8, &sm.s.a[buf][ar][ak + 8]);
    const __bf16* gb = Bm + (size_t)(kk + bk) * Ncols + n0 + bn;
    cp16_g2l(gb, &sm.s.b[buf][bk][bn]);
    cp16_g2l(gb + 8, &sm.s.b[buf][bk][bn + 8]);
  };

  v8f acc[2][4];
#pragma unroll
  for (int i = 0; i < 2; ++i)
#pragma unroll
    for (int j = 0; j < 4; ++j)
#pragma unroll
      for (int e = 0; e < 8; ++e) acc[i][j][e] = 0.0f;

  stage(0, 0);
  wait_async();
  __syncthreads();
  int buf = 0;
  for (int kk = 0; kk < Kdim; kk += 32) {
    if (kk + 32 < Kdim) stage(buf ^ 1, kk + 32);  // overlap DMA with WMMA

    v16bf af0 = frag_a(&sm.s.a[buf][wm * 32][0], 40, lane);
    v16bf af1 = frag_a(&sm.s.a[buf][wm * 32 + 16][0], 40, lane);
    v16bf bfj[4];
#pragma unroll
    for (int j = 0; j < 4; ++j) bfj[j] = frag_b(&sm.s.b[buf][0][wn * 64 + j * 16], 136, lane);
#pragma unroll
    for (int j = 0; j < 4; ++j) {
      acc[0][j] = wmma_bf16(af0, bfj[j], acc[0][j]);
      acc[1][j] = wmma_bf16(af1, bfj[j], acc[1][j]);
    }
    wait_async();
    __syncthreads();
    buf ^= 1;
  }

  if constexpr (EPI == 1) {
#pragma unroll
    for (int i = 0; i < 2; ++i)
#pragma unroll
      for (int j = 0; j < 4; ++j)
#pragma unroll
        for (int e = 0; e < 8; ++e) {
          int r = m0 + wm * 32 + i * 16 + e + hi8;
          int c = n0 + wn * 64 + j * 16 + nlo;
          Out[(size_t)r * Ncols + c] = acc[i][j][e];
        }
  } else {
    // accum tile -> LDS, RoPE, scatter bf16 head-major [B*H][T][128]
#pragma unroll
    for (int i = 0; i < 2; ++i)
#pragma unroll
      for (int j = 0; j < 4; ++j)
#pragma unroll
        for (int e = 0; e < 8; ++e)
          sm.c[wm * 32 + i * 16 + e + hi8][wn * 64 + j * 16 + nlo] = acc[i][j][e];
    __syncthreads();

    const int sec = n0 >> 11;         // 0=Q 1=K 2=V
    const int hh = (n0 & 2047) >> 7;  // head
    __bf16* dst = (sec == 0) ? Qh : (sec == 1) ? Kh : Vh;
    const int r = tid >> 1;
    const int dbase = (tid & 1) << 6;
    const int gm = m0 + r;
    const int bb = gm >> 11, tt = gm & 2047;
    const size_t base = ((size_t)(bb * 16 + hh) * 2048 + tt) * 128;
#pragma unroll 8
    for (int d0 = 0; d0 < 64; ++d0) {
      int d = dbase + d0;
      float v;
      if (sec == 2) {
        v = sm.c[r][d];
      } else {
        int idx = d & 63;
        float ang = (float)tt * __expf(-(float)idx * 0.14391156831212787f);  // ln(1e4)/64
        float sv, cv;
        __sincosf(ang, &sv, &cv);
        v = (d < 64) ? sm.c[r][d] * cv - sm.c[r][d + 64] * sv
                     : sm.c[r][d] * cv + sm.c[r][d - 64] * sv;
      }
      dst[base + d] = (__bf16)v;
    }
  }
}

// ---------- kernel 3: block-cooperative flash attention ----------
// Q/K/V: [B*H][T=2048][Dh=128] bf16. Block = 128 query rows; waves share
// async-staged K/V tiles (32 keys) in LDS; each wave owns 16 rows.
__global__ __launch_bounds__(256) void flash_attn(const __bf16* __restrict__ Q,
                                                  const __bf16* __restrict__ K,
                                                  const __bf16* __restrict__ V,
                                                  __bf16* __restrict__ Oa) {
  __shared__ __bf16 sK[2][32][136];
  __shared__ __bf16 sV[2][32][136];
  __shared__ __bf16 sP[8][16 * 32];
  const int tid = threadIdx.x;
  const int lane = tid & 31;
  const int w = tid >> 5;
  const int bh = blockIdx.y;
  const int b = bh >> 4, h = bh & 15;
  const int r0 = blockIdx.x * 128;
  const int q0 = r0 + w * 16;
  const int hi8 = (lane & 16) ? 8 : 0;
  const int nlo = lane & 15;
  const float scale = 0.08838834764831845f;  // 1/sqrt(128)

  const __bf16* Qh = Q + (size_t)bh * (2048 * 128);
  const __bf16* Kh = K + (size_t)bh * (2048 * 128);
  const __bf16* Vh = V + (size_t)bh * (2048 * 128);

  const int srow = tid >> 3;        // 0..31 (key within tile)
  const int scol = (tid & 7) << 4;  // 0..112 (dh)
  auto stage = [&](int buf, int kt) {
    const __bf16* gk = Kh + ((size_t)kt * 32 + srow) * 128 + scol;
    const __bf16* gv = Vh + ((size_t)kt * 32 + srow) * 128 + scol;
    cp16_g2l(gk, &sK[buf][srow][scol]);
    cp16_g2l(gk + 8, &sK[buf][srow][scol + 8]);
    cp16_g2l(gv, &sV[buf][srow][scol]);
    cp16_g2l(gv + 8, &sV[buf][srow][scol + 8]);
  };

  v16bf qf[4];
#pragma unroll
  for (int kd = 0; kd < 4; ++kd) qf[kd] = frag_a(Qh + (size_t)q0 * 128 + kd * 32, 128, lane);

  v8f o[8];
  float m[8], l[8];
#pragma unroll
  for (int t = 0; t < 8; ++t)
#pragma unroll
    for (int e = 0; e < 8; ++e) o[t][e] = 0.0f;
#pragma unroll
  for (int i = 0; i < 8; ++i) { m[i] = -3.0e38f; l[i] = 0.0f; }

  const int ktiles = r0 / 32 + 4;  // uniform across block -> barriers are safe
  stage(0, 0);
  wait_async();
  __syncthreads();
  int buf = 0;
  for (int kt = 0; kt < ktiles; ++kt) {
    if (kt + 1 < ktiles) stage(buf ^ 1, kt + 1);  // overlap DMA with WMMA
    const int key0 = kt * 32;

    if (q0 + 15 >= key0) {  // skip fully-masked tiles (barrier parity kept)
      // S = Q * K^T : batched fragment loads, then 8 WMMAs
      v16bf kf[2][4];
#pragma unroll
      for (int n = 0; n < 2; ++n)
#pragma unroll
        for (int kd = 0; kd < 4; ++kd)
          kf[n][kd] = frag_bT(&sK[buf][n * 16][kd * 32], 136, lane);
      v8f s[2];
#pragma unroll
      for (int n = 0; n < 2; ++n) {
#pragma unroll
        for (int e = 0; e < 8; ++e) s[n][e] = 0.0f;
#pragma unroll
        for (int kd = 0; kd < 4; ++kd) s[n] = wmma_bf16(qf[kd], kf[n][kd], s[n]);
      }
      // scale + causal mask + per-row tile max
      float tmax[8];
#pragma unroll
      for (int i = 0; i < 8; ++i) tmax[i] = -3.0e38f;
#pragma unroll
      for (int n = 0; n < 2; ++n)
#pragma unroll
        for (int i = 0; i < 8; ++i) {
          int row = q0 + i + hi8;
          int col = key0 + n * 16 + nlo;
          float v = s[n][i] * scale;
          if (col > row) v = -3.0e38f;
          s[n][i] = v;
          tmax[i] = fmaxf(tmax[i], v);
        }
      // online softmax (reduce across 16-lane half-wave matching C layout)
#pragma unroll
      for (int i = 0; i < 8; ++i) {
#pragma unroll
        for (int d = 1; d < 16; d <<= 1) tmax[i] = fmaxf(tmax[i], __shfl_xor(tmax[i], d, 32));
        float mn = fmaxf(m[i], tmax[i]);
        float corr = __expf(m[i] - mn);
        m[i] = mn;
        l[i] *= corr;
#pragma unroll
        for (int t = 0; t < 8; ++t) o[t][i] *= corr;
      }
      float rsum[8];
#pragma unroll
      for (int i = 0; i < 8; ++i) rsum[i] = 0.0f;
#pragma unroll
      for (int n = 0; n < 2; ++n)
#pragma unroll
        for (int i = 0; i < 8; ++i) {
          float p = __expf(s[n][i] - m[i]);
          rsum[i] += p;
          sP[w][(i + hi8) * 32 + n * 16 + nlo] = (__bf16)p;  // C-layout -> A-layout
        }
#pragma unroll
      for (int i = 0; i < 8; ++i) {
#pragma unroll
        for (int d = 1; d < 16; d <<= 1) rsum[i] += __shfl_xor(rsum[i], d, 32);
        l[i] += rsum[i];
      }
      lds_fence();  // wave-local P handoff through LDS
      v16bf pf = frag_a(&sP[w][0], 32, lane);
      v16bf vf[8];
#pragma unroll
      for (int t = 0; t < 8; ++t) vf[t] = frag_b(&sV[buf][0][t * 16], 136, lane);
#pragma unroll
      for (int t = 0; t < 8; ++t) o[t] = wmma_bf16(pf, vf[t], o[t]);
    }

    wait_async();
    __syncthreads();
    buf ^= 1;
  }

  // normalize + store bf16 to attn buffer [B][T][C], C = h*128+dh
  float invl[8];
#pragma unroll
  for (int i = 0; i < 8; ++i) invl[i] = 1.0f / l[i];
#pragma unroll
  for (int t = 0; t < 8; ++t)
#pragma unroll
    for (int i = 0; i < 8; ++i) {
      int trow = q0 + i + hi8;
      int col = h * 128 + t * 16 + nlo;
      Oa[((size_t)(b * 2048 + trow)) * 2048 + col] = (__bf16)(o[t][i] * invl[i]);
    }
}

// ---------- host launcher ----------
extern "C" void kernel_launch(void* const* d_in, const int* in_sizes, int n_in,
                              void* d_out, int out_size, void* d_ws, size_t ws_size,
                              hipStream_t stream) {
  const float* x = (const float*)d_in[0];      // [2,2048,2048]
  const float* wqkv = (const float*)d_in[1];   // [2048,6144]
  const float* wproj = (const float*)d_in[2];  // [2048,2048]
  float* out = (float*)d_out;                  // [2,2048,2048]

  const size_t NX = 8388608, NWQ = 12582912, NWP = 4194304, NH = 8388608;
  __bf16* xb = (__bf16*)d_ws;
  __bf16* wqkvb = xb + NX;
  __bf16* wprojb = wqkvb + NWQ;
  __bf16* Qb = wprojb + NWP;
  __bf16* Kb = Qb + NH;
  __bf16* Vb = Kb + NH;
  __bf16* attnb = Vb + NH;  // ~112 MB total workspace use

  cvt_f32_bf16<<<2048, 256, 0, stream>>>(x, xb, (int)NX);
  cvt_f32_bf16<<<2048, 256, 0, stream>>>(wqkv, wqkvb, (int)NWQ);
  cvt_f32_bf16<<<2048, 256, 0, stream>>>(wproj, wprojb, (int)NWP);

  gemm_bf16<0><<<dim3(48, 32), 256, 0, stream>>>(xb, wqkvb, 2048, 6144, Qb, Kb, Vb, nullptr);
  flash_attn<<<dim3(16, 32), 256, 0, stream>>>(Qb, Kb, Vb, attnb);
  gemm_bf16<1><<<dim3(16, 32), 256, 0, stream>>>(attnb, wprojb, 2048, 2048,
                                                 nullptr, nullptr, nullptr, out);
}
